// ConvPosMultiHeadAttn_1786706395828
// MI455X (gfx1250) — matrix operand order repaired and checked by reading, not compiled, not run on hardware
//
#include <hip/hip_runtime.h>

// ---------------------------------------------------------------------------
// Types for gfx1250 WMMA (wave32)
// ---------------------------------------------------------------------------
typedef __bf16 bf16_t;
typedef __attribute__((ext_vector_type(16))) __bf16 v16bf;
typedef __attribute__((ext_vector_type(8)))  __bf16 v8bf;
typedef __attribute__((ext_vector_type(4)))  __bf16 v4bf;
typedef __attribute__((ext_vector_type(8)))  float  v8f;

#define WMMA_BF16(a, b, c) \
  __builtin_amdgcn_wmma_f32_16x16x32_bf16(false, (a), false, (b), (short)0, (c), false, false)

// Problem constants (from the reference)
constexpr int Bc  = 8;
constexpr int Lc  = 512;
constexpr int Dc  = 1024;
constexpr int Hc  = 16;
constexpr int HDc = 64;
constexpr int MLc = Bc * Lc;      // 4096 flattened rows
constexpr int QKV_LD = 5 * Dc;    // 5120
constexpr int POS_LD = 3 * Dc;    // 3072

// ---------------------------------------------------------------------------
// WMMA operand loaders (same contiguous k order for A and B, wave32)
//   A (16x32 bf16): lane row = lane&15; K = hi*16 .. hi*16+15 (contiguous)
//   B (32x16 bf16): lane col = lane&15; K = hi*16 .. hi*16+15 (contiguous)
// ---------------------------------------------------------------------------
__device__ __forceinline__ v16bf loadA(const bf16_t* base, int stride) {
  const int lane = threadIdx.x & 31;
  const bf16_t* p = base + (size_t)(lane & 15) * stride + ((lane >> 4) << 4);
  union { v16bf v; v8bf h[2]; } u;
  u.h[0] = *(const v8bf*)(p);
  u.h[1] = *(const v8bf*)(p + 8);
  return u.v;
}

__device__ __forceinline__ v16bf loadB(const bf16_t* base, int stride) {
  const int lane = threadIdx.x & 31;
  const bf16_t* p = base + (size_t)(lane & 15) * stride + ((lane >> 4) << 4);
  union { v16bf v; v8bf h[2]; } u;
  u.h[0] = *(const v8bf*)(p);
  u.h[1] = *(const v8bf*)(p + 8);
  return u.v;
}

// Augmented 128-feature operand: ks=0,1 -> batch qkv slice; ks=2,3 -> pos slice
__device__ __forceinline__ const bf16_t* aug_ptr(const bf16_t* qkvb, const bf16_t* posb,
                                                 int b, int h, int row, int ks,
                                                 int boff, int poff, int inner) {
  if (ks < 2)
    return qkvb + (size_t)(b * Lc + row) * QKV_LD + boff + h * HDc + ks * 32 + inner;
  return posb + (size_t)row * POS_LD + poff + h * HDc + (ks - 2) * 32 + inner;
}

__device__ __forceinline__ v16bf load_augA(const bf16_t* qkvb, const bf16_t* posb,
                                           int b, int h, int row0, int ks, int boff, int poff) {
  const int lane = threadIdx.x & 31;
  const bf16_t* p = aug_ptr(qkvb, posb, b, h, row0 + (lane & 15), ks, boff, poff,
                            (lane >> 4) << 4);
  union { v16bf v; v8bf h2[2]; } u;
  u.h2[0] = *(const v8bf*)(p);
  u.h2[1] = *(const v8bf*)(p + 8);
  return u.v;
}

__device__ __forceinline__ v16bf load_augB(const bf16_t* qkvb, const bf16_t* posb,
                                           int b, int h, int row0, int ks, int boff, int poff) {
  const int lane = threadIdx.x & 31;
  const bf16_t* p = aug_ptr(qkvb, posb, b, h, row0 + (lane & 15), ks, boff, poff,
                            (lane >> 4) << 4);
  union { v16bf v; v8bf h2[2]; } u;
  u.h2[0] = *(const v8bf*)(p);
  u.h2[1] = *(const v8bf*)(p + 8);
  return u.v;
}

// ---------------------------------------------------------------------------
// fp32 -> bf16 conversion (x4 vectorized)
// ---------------------------------------------------------------------------
__global__ __launch_bounds__(256) void cvt_f32_to_bf16(const float* __restrict__ in,
                                                       bf16_t* __restrict__ out, int n4) {
  const int i = blockIdx.x * 256 + threadIdx.x;
  if (i < n4) {
    const float4 v = ((const float4*)in)[i];
    v4bf o;
    o[0] = (bf16_t)v.x; o[1] = (bf16_t)v.y; o[2] = (bf16_t)v.z; o[3] = (bf16_t)v.w;
    ((v4bf*)out)[i] = o;
  }
}

// ---------------------------------------------------------------------------
// Positional projection: y[l, e] = sum_d pe[l, d] * W_pos[e, d]  -> bf16
// ---------------------------------------------------------------------------
__global__ __launch_bounds__(256) void pos_embed_kernel(const float* __restrict__ wpos,
                                                        bf16_t* __restrict__ posb) {
  __shared__ float pe[64];
  const int l = blockIdx.x;
  const int tid = threadIdx.x;
  if (tid < 64) {
    const int f = tid & 31;
    const float invf = expf(-(float)f * (9.210340371976184f / 31.0f)); // ln(1e4)/31
    const float ang = (float)(l - 256) * invf;
    pe[tid] = (tid < 32) ? sinf(ang) : cosf(ang);
  }
  __syncthreads();
  for (int e = tid; e < POS_LD; e += 256) {
    const float* wr = wpos + (size_t)e * 64;
    float s = 0.f;
#pragma unroll 16
    for (int d = 0; d < 64; ++d) s += pe[d] * wr[d];
    posb[(size_t)l * POS_LD + e] = (bf16_t)s;
  }
}

// ---------------------------------------------------------------------------
// QKV projection: qkvb[ML, 5120] = xb[ML, 1024] @ W_qkv^T (bf16 out).
// Dead 'k' slice (cols 1024..2047) skipped: 256 live N-tiles of 16.
// Ping-pong double-buffered K loop, fully unrolled: no register rotation
// copies, next k-step tiles always in flight during current WMMAs.
// ---------------------------------------------------------------------------
__global__ __launch_bounds__(256) void gemm_qkv_kernel(const bf16_t* __restrict__ xb,
                                                       const bf16_t* __restrict__ wb,
                                                       bf16_t* __restrict__ qkvb) {
  const int w = threadIdx.x >> 5, lane = threadIdx.x & 31;
  const int m0 = blockIdx.x * 32 + (w & 1) * 16;
  const int tg = blockIdx.y * 16 + (w >> 1) * 4;
  int col0[4];
#pragma unroll
  for (int q = 0; q < 4; ++q) {
    const int t = tg + q;
    col0[q] = (t < 64 ? t : t + 64) << 4;   // skip k slice [1024,2048)
  }
  v8f acc[4] = {};
  const bf16_t* arow = xb + (size_t)m0 * Dc;

  // prologue: stage k = 0 into buffer 0
  v16bf a0 = loadA(arow, Dc);
  v16bf b0[4];
#pragma unroll
  for (int q = 0; q < 4; ++q) b0[q] = loadB(wb + (size_t)col0[q] * Dc, Dc);

#pragma unroll
  for (int k0 = 0; k0 < Dc; k0 += 64) {
    // stage k0+32 into buffer 1 (overlaps buffer-0 WMMAs below)
    v16bf a1 = loadA(arow + k0 + 32, Dc);
    v16bf b1[4];
#pragma unroll
    for (int q = 0; q < 4; ++q) b1[q] = loadB(wb + (size_t)col0[q] * Dc + k0 + 32, Dc);
#pragma unroll
    for (int q = 0; q < 4; ++q) acc[q] = WMMA_BF16(a0, b0[q], acc[q]);
    if (k0 + 64 < Dc) {   // folds at compile time under full unroll
      a0 = loadA(arow + k0 + 64, Dc);
#pragma unroll
      for (int q = 0; q < 4; ++q) b0[q] = loadB(wb + (size_t)col0[q] * Dc + k0 + 64, Dc);
    }
#pragma unroll
    for (int q = 0; q < 4; ++q) acc[q] = WMMA_BF16(a1, b1[q], acc[q]);
  }

  const int n = lane & 15, hi = lane >> 4;
#pragma unroll
  for (int q = 0; q < 4; ++q)
#pragma unroll
    for (int r = 0; r < 8; ++r)
      qkvb[(size_t)(m0 + r + 8 * hi) * QKV_LD + col0[q] + n] = (bf16_t)acc[q][r];
}

// ---------------------------------------------------------------------------
// V transpose: Vt[b][h][d][j] = qkv.v[b, j, h*64+d]   (bf16 -> bf16)
// ---------------------------------------------------------------------------
__global__ __launch_bounds__(256) void transpose_v_kernel(const bf16_t* __restrict__ qkvb,
                                                          bf16_t* __restrict__ vt) {
  const int idx = blockIdx.x * 256 + threadIdx.x;   // 0 .. B*H*64*512-1
  const int j  = idx & 511;
  const int d  = (idx >> 9) & 63;
  const int bh = idx >> 15;
  vt[idx] = qkvb[(size_t)((bh >> 4) * Lc + j) * QKV_LD + 4 * Dc + (bh & 15) * HDc + d];
}

// ---------------------------------------------------------------------------
// Combined score tile (16x16): all 8 B-tile loads issued before the 8 WMMAs.
// ---------------------------------------------------------------------------
__device__ __forceinline__ v8f score_tile(const bf16_t* qkvb, const bf16_t* posb,
                                          const v16bf* aQ, int b, int h, int jt,
                                          const int* smq, const int* smm,
                                          const int* qmi, int i0w,
                                          float shift, float bias) {
  v16bf b1[4], b2[4];
#pragma unroll
  for (int ks = 0; ks < 4; ++ks) {
    b1[ks] = load_augB(qkvb, posb, b, h, jt * 16, ks, 2 * Dc, 1 * Dc); // k1 | k1_p
    b2[ks] = load_augB(qkvb, posb, b, h, jt * 16, ks, 3 * Dc, 2 * Dc); // k2 | k2_p
  }
  v8f ss = {};
  v8f so = {};
#pragma unroll
  for (int ks = 0; ks < 4; ++ks) {
    ss = WMMA_BF16(aQ[ks], b1[ks], ss);
    so = WMMA_BF16(aQ[ks], b2[ks], so);
  }
  const int lane = threadIdx.x & 31;
  const int n = lane & 15, hi = lane >> 4;
  const int j = jt * 16 + n;
  const int qmj = smq[j];
  const bool masked = (smm[j] == 0);
  v8f out;
#pragma unroll
  for (int r = 0; r < 8; ++r) {
    const int i = i0w + r + 8 * hi;
    float s = (qmi[r] == qmj) ? ss[r] : so[r];
    if (masked) s = -1e30f;
    const float dd = (float)(i - j);
    out[r] = s - (shift * dd * dd + bias);
  }
  return out;
}

// ---------------------------------------------------------------------------
// Attention: block = 4 waves, one (b,h) + 64 query rows per block.
// Two-pass softmax with score recompute; P@V through a per-wave LDS staging
// tile that converts C-layout scores into the A-layout bf16 WMMA operand.
// ---------------------------------------------------------------------------
__global__ __launch_bounds__(128) void attn_kernel(const bf16_t* __restrict__ qkvb,
                                                   const bf16_t* __restrict__ posb,
                                                   const bf16_t* __restrict__ vt,
                                                   const int* __restrict__ mask,
                                                   const int* __restrict__ qmask,
                                                   const float* __restrict__ shiftp,
                                                   const float* __restrict__ biasp,
                                                   bf16_t* __restrict__ obuf) {
  __shared__ int   smq[Lc];
  __shared__ int   smm[Lc];
  __shared__ float ps[4][16][32];   // per-wave P staging (C-layout -> A-layout)

  const int bh = blockIdx.x;
  const int b = bh >> 4, h = bh & 15;
  const int w = threadIdx.x >> 5, lane = threadIdx.x & 31;
  const int i0w = blockIdx.y * 64 + w * 16;
  const int n = lane & 15, hi = lane >> 4;

  for (int t = threadIdx.x; t < Lc; t += 128) {
    smq[t] = qmask[b * Lc + t];
    smm[t] = mask[b * Lc + t];
  }
  __syncthreads();

  const float shift = shiftp[0];
  const float bias = biasp[0];

  int qmi[8];
#pragma unroll
  for (int r = 0; r < 8; ++r) qmi[r] = smq[i0w + r + 8 * hi];

  // Augmented Q operand (q | q_p), kept in registers across both passes
  v16bf aQ[4];
#pragma unroll
  for (int ks = 0; ks < 4; ++ks) aQ[ks] = load_augA(qkvb, posb, b, h, i0w, ks, 0, 0);

  // ---- pass 1: row maxima -------------------------------------------------
  float rmax[8];
#pragma unroll
  for (int r = 0; r < 8; ++r) rmax[r] = -3.0e38f;
#pragma unroll 2
  for (int jt = 0; jt < Lc / 16; ++jt) {
    v8f s = score_tile(qkvb, posb, aQ, b, h, jt, smq, smm, qmi, i0w, shift, bias);
#pragma unroll
    for (int r = 0; r < 8; ++r) rmax[r] = fmaxf(rmax[r], s[r]);
  }
#pragma unroll
  for (int m = 1; m < 16; m <<= 1)
#pragma unroll
    for (int r = 0; r < 8; ++r)
      rmax[r] = fmaxf(rmax[r], __shfl_xor(rmax[r], m, 32));

  // ---- pass 2: exp, row sums, and P@V -------------------------------------
  float rsum[8] = {0.f, 0.f, 0.f, 0.f, 0.f, 0.f, 0.f, 0.f};
  v8f acc[4] = {};
  for (int js = 0; js < Lc / 32; ++js) {
#pragma unroll
    for (int sub = 0; sub < 2; ++sub) {
      const int jt = js * 2 + sub;
      v8f s = score_tile(qkvb, posb, aQ, b, h, jt, smq, smm, qmi, i0w, shift, bias);
#pragma unroll
      for (int r = 0; r < 8; ++r) {
        const float p = __expf(s[r] - rmax[r]);
        rsum[r] += p;
        ps[w][r + 8 * hi][n + 16 * sub] = p;
      }
    }
    __syncthreads();
    // B tiles from Vt first (independent of LDS), then the A-layout P tile
    union { v16bf v; v8bf h2[2]; } vb[4];
#pragma unroll
    for (int nt = 0; nt < 4; ++nt) {
      const int d = nt * 16 + n;
      const bf16_t* vb_p = vt + ((size_t)(bh * HDc + d) * Lc) + js * 32 + hi * 16;
      vb[nt].h2[0] = *(const v8bf*)(vb_p);
      vb[nt].h2[1] = *(const v8bf*)(vb_p + 8);
    }
    union { v16bf v; bf16_t e[16]; } pa;
    {
      const int kb = hi * 16;
      const float* prow = &ps[w][lane & 15][0];
#pragma unroll
      for (int e2 = 0; e2 < 8; ++e2) {
        pa.e[e2]     = (bf16_t)prow[kb + e2];
        pa.e[e2 + 8] = (bf16_t)prow[kb + 8 + e2];
      }
    }
#pragma unroll
    for (int nt = 0; nt < 4; ++nt)
      acc[nt] = WMMA_BF16(pa.v, vb[nt].v, acc[nt]);
    __syncthreads();
  }
#pragma unroll
  for (int m = 1; m < 16; m <<= 1)
#pragma unroll
    for (int r = 0; r < 8; ++r)
      rsum[r] += __shfl_xor(rsum[r], m, 32);

  // ---- epilogue: normalize and emit bf16 attention output -----------------
#pragma unroll
  for (int r = 0; r < 8; ++r) {
    const float inv = 1.0f / rsum[r];
    const size_t row = (size_t)(b * Lc + i0w + r + 8 * hi);
#pragma unroll
    for (int nt = 0; nt < 4; ++nt)
      obuf[row * Dc + h * HDc + nt * 16 + n] = (bf16_t)(acc[nt][r] * inv);
  }
}

// ---------------------------------------------------------------------------
// Output projection: out[ML, 1024] = obuf @ W_fc^T + b_fc   (f32 out),
// same ping-pong double-buffered K loop as the QKV GEMM.
// ---------------------------------------------------------------------------
__global__ __launch_bounds__(256) void gemm_fc_kernel(const bf16_t* __restrict__ ob,
                                                      const bf16_t* __restrict__ wfc,
                                                      const float* __restrict__ bfc,
                                                      float* __restrict__ out) {
  const int w = threadIdx.x >> 5, lane = threadIdx.x & 31;
  const int m0 = blockIdx.x * 32 + (w & 1) * 16;
  const int t0 = blockIdx.y * 16 + (w >> 1) * 4;
  v8f acc[4] = {};
  const bf16_t* arow = ob + (size_t)m0 * Dc;

  v16bf a0 = loadA(arow, Dc);
  v16bf b0[4];
#pragma unroll
  for (int q = 0; q < 4; ++q) b0[q] = loadB(wfc + (size_t)(t0 + q) * 16 * Dc, Dc);

#pragma unroll
  for (int k0 = 0; k0 < Dc; k0 += 64) {
    v16bf a1 = loadA(arow + k0 + 32, Dc);
    v16bf b1[4];
#pragma unroll
    for (int q = 0; q < 4; ++q) b1[q] = loadB(wfc + (size_t)(t0 + q) * 16 * Dc + k0 + 32, Dc);
#pragma unroll
    for (int q = 0; q < 4; ++q) acc[q] = WMMA_BF16(a0, b0[q], acc[q]);
    if (k0 + 64 < Dc) {
      a0 = loadA(arow + k0 + 64, Dc);
#pragma unroll
      for (int q = 0; q < 4; ++q) b0[q] = loadB(wfc + (size_t)(t0 + q) * 16 * Dc + k0 + 64, Dc);
    }
#pragma unroll
    for (int q = 0; q < 4; ++q) acc[q] = WMMA_BF16(a1, b1[q], acc[q]);
  }

  const int n = lane & 15, hi = lane >> 4;
#pragma unroll
  for (int q = 0; q < 4; ++q) {
    const int col = (t0 + q) * 16 + n;
    const float bv = bfc[col];
#pragma unroll
    for (int r = 0; r < 8; ++r)
      out[(size_t)(m0 + r + 8 * hi) * Dc + col] = acc[q][r] + bv;
  }
}

// ---------------------------------------------------------------------------
// Host launcher
// ---------------------------------------------------------------------------
extern "C" void kernel_launch(void* const* d_in, const int* in_sizes, int n_in,
                              void* d_out, int out_size, void* d_ws, size_t ws_size,
                              hipStream_t stream) {
  (void)in_sizes; (void)n_in; (void)out_size; (void)ws_size;

  const float* x      = (const float*)d_in[0];   // [B, L, D]
  const int*   mask   = (const int*)d_in[1];     // [B, L]
  const int*   qmask  = (const int*)d_in[2];     // [B, L]
  const float* Wqkv   = (const float*)d_in[3];   // [5D, D]
  const float* Wpos   = (const float*)d_in[4];   // [3D, 64]
  const float* Wfc    = (const float*)d_in[5];   // [D, D]
  const float* bfc    = (const float*)d_in[6];   // [D]
  const float* shiftp = (const float*)d_in[7];   // [1]
  const float* biasp  = (const float*)d_in[8];   // [1]
  float* out = (float*)d_out;                    // [B, L, D] f32

  // Workspace partition (all sizes 256B-aligned); total ~83 MB
  char* ws = (char*)d_ws;
  size_t off = 0;
  auto take = [&](size_t bytes) { char* p = ws + off; off += (bytes + 255) & ~(size_t)255; return p; };
  bf16_t* xb    = (bf16_t*)take((size_t)MLc * Dc * 2);        // x bf16
  bf16_t* wqkvb = (bf16_t*)take((size_t)5 * Dc * Dc * 2);     // W_qkv bf16
  bf16_t* wfcb  = (bf16_t*)take((size_t)Dc * Dc * 2);         // W_fc bf16
  bf16_t* qkvb  = (bf16_t*)take((size_t)MLc * QKV_LD * 2);    // qkv bf16 (k slice unused)
  bf16_t* posb  = (bf16_t*)take((size_t)Lc * POS_LD * 2);     // pos proj bf16
  bf16_t* vt    = (bf16_t*)take((size_t)Bc * Hc * HDc * Lc * 2); // V transposed
  bf16_t* obuf  = (bf16_t*)take((size_t)MLc * Dc * 2);        // attention output bf16

  // 1) fp32 -> bf16 conversions
  cvt_f32_to_bf16<<<(MLc * Dc / 4 + 255) / 256, 256, 0, stream>>>(x, xb, MLc * Dc / 4);
  cvt_f32_to_bf16<<<(5 * Dc * Dc / 4 + 255) / 256, 256, 0, stream>>>(Wqkv, wqkvb, 5 * Dc * Dc / 4);
  cvt_f32_to_bf16<<<(Dc * Dc / 4 + 255) / 256, 256, 0, stream>>>(Wfc, wfcb, Dc * Dc / 4);

  // 2) positional projection (tiny)
  pos_embed_kernel<<<Lc, 256, 0, stream>>>(Wpos, posb);

  // 3) QKV projection (skips dead 'k' slice): grid 128 x 16, 8 waves/block
  gemm_qkv_kernel<<<dim3(MLc / 32, 16), 256, 0, stream>>>(xb, wqkvb, qkvb);

  // 4) V transpose for the P@V B operand
  transpose_v_kernel<<<(Bc * Hc * HDc * Lc) / 256, 256, 0, stream>>>(qkvb, vt);

  // 5) attention: one (b,h) x 64-query block per workgroup
  attn_kernel<<<dim3(Bc * Hc, Lc / 64), 128, 0, stream>>>(qkvb, posb, vt, mask, qmask,
                                                          shiftp, biasp, obuf);

  // 6) output projection + bias -> f32
  gemm_fc_kernel<<<dim3(MLc / 32, Dc / 256), 256, 0, stream>>>(obuf, wfcb, bfc, out);
}
